// DepthLSSTransform_Seg_42107859370545
// MI455X (gfx1250) — compile-verified, matrix operand order checked by reading, not run on hardware
//
#include <hip/hip_runtime.h>
#include <hip/hip_bf16.h>

// ---------------------------------------------------------------------------
// DepthLSSTransform for gfx1250 (MI455X):
//  - heavy convs = implicit GEMM on V_WMMA_F32_16X16X32_BF16 (wave32)
//  - per-tap weight tiles staged LDS-side by the Tensor Data Mover
//    (tensor_load_to_lds + s_wait_tensorcnt)
//  - bev_pool via fp32 global atomics (41 MB accumulator is L2-resident)
// ---------------------------------------------------------------------------

typedef __attribute__((ext_vector_type(8)))  __bf16 v8bf_t;
typedef __attribute__((ext_vector_type(16))) __bf16 v16bf_t;
typedef __attribute__((ext_vector_type(8)))  float  v8f_t;
typedef __attribute__((ext_vector_type(4)))  unsigned int v4u_t;
typedef __attribute__((ext_vector_type(8)))  int v8i_t;
typedef __attribute__((ext_vector_type(4)))  int v4i_t;

#if defined(__has_builtin)
#  if __has_builtin(__builtin_amdgcn_tensor_load_to_lds)
#    define HAVE_TDM 1
#  else
#    define HAVE_TDM 0
#  endif
#else
#  define HAVE_TDM 0
#endif

#define BN_   6
#define CIN_  256
#define CIMG_ 80
#define DD_   59
#define FH_   32
#define FW_   88
#define IH_   256
#define IW_   704
#define NXG_  360
#define NPIX_ (BN_*FH_*FW_)        /* 16896 */
#define NPTS_ (BN_*DD_*FH_*FW_)    /* 996864 */

static __device__ __forceinline__ __bf16 f2bf(float f) {
  unsigned u = __builtin_bit_cast(unsigned, f);
  u += 0x7FFFu + ((u >> 16) & 1u);                 // round to nearest even
  unsigned short h = (unsigned short)(u >> 16);
  return __builtin_bit_cast(__bf16, h);
}

// ---------------------------------------------------------------------------
// Implicit-GEMM conv. X pixel-major bf16 [img*H*W][CINP]; weights pre-packed
// per-tap as [tap][O_pad][CINP] bf16. Block = 256 thr = 8 waves laid out
// 4(M) x 2(N); each wave owns a 16Mx32N strip (2 accumulators, shared A frag)
// -> block tile 64M x 64N. Per tap: TDM stages the 64 x CINP weight plane
// into LDS once; K-loop stages 64x32 im2col A tiles cooperatively.
// SMODE: 0 = bf16 pixel-major out, 1 = f32 pixel-major out, 2 = f32 NCHW out.
// flags: 1 = bias, 2 = folded-BN scale/shift, 4 = relu.
// ---------------------------------------------------------------------------
template<int SMODE, int CINP>
__global__ __launch_bounds__(256) void conv_wmma(
    const __bf16* __restrict__ Xin, const __bf16* __restrict__ Wt,
    void* __restrict__ Out,
    const float* __restrict__ bias, const float* __restrict__ bns,
    const float* __restrict__ bnt, int flags,
    int IHd, int IWd, int OHd, int OWd, int KH, int KW, int strd, int pad,
    int O_pad, int Nmax, int strideN, int M)
{
  __shared__ __bf16 As[64][32];
  __shared__ __bf16 Bs[64][CINP];
  const int t    = threadIdx.x;
  const int lane = t & 31;
  const int wv   = t >> 5;
  const int mi   = wv >> 1;            // 0..3
  const int ni   = wv & 1;             // 0..1
  const int m_blk = blockIdx.x * 64;
  const int n_blk = blockIdx.y * 64;
  const int OHW = OHd * OWd;

  // A-tile duty: 64 rows x 32 k, 16B (8 bf16) per thread
  const int arow = t >> 2;
  const int akq  = (t & 3) << 3;
  const int am   = m_blk + arow;
  int aimg = 0, aoh = 0, aow = 0;
  const bool am_ok = (am < M);
  if (am_ok) { aimg = am / OHW; int r = am - aimg * OHW; aoh = r / OWd; aow = r - aoh * OWd; }

  const int half = lane >> 4;
  const int lr   = lane & 15;
  const size_t btap = (size_t)O_pad * CINP;

  v8f_t acc0 = {}, acc1 = {};
  const int ntap = KH * KW;
  for (int tap = 0; tap < ntap; ++tap) {
    // ---- stage this tap's 64 x CINP weight plane into LDS ----
    __syncthreads();                       // everyone done reading prior tap
#if HAVE_TDM
    if (wv == 0) {
      const __bf16* wsrc = Wt + btap * tap + (size_t)n_blk * CINP;
      unsigned long long ga = (unsigned long long)(size_t)wsrc;
      unsigned ldsb = (unsigned)(size_t)(void*)&Bs[0][0];
      const unsigned td0 = (unsigned)(CINP / 2);   // dwords per row
      v4u_t g0;
      g0[0] = 1u;                                  // count=1, user descriptor
      g0[1] = ldsb;                                // lds_addr
      g0[2] = (unsigned)ga;                        // global_addr[31:0]
      g0[3] = (unsigned)((ga >> 32) & 0x01FFFFFFull) | (2u << 30); // [56:32]|type=2
      v8i_t g1;
      g1[0] = (int)(2u << 16);                     // data_size = 4B
      g1[1] = (int)((td0 & 0xFFFFu) << 16);        // tensor_dim0 lo16
      g1[2] = (int)((td0 >> 16) | (64u << 16));    // tensor_dim0 hi16 | tensor_dim1 lo16
      g1[3] = (int)((td0 & 0xFFFFu) << 16);        // tensor_dim1 hi16=0 | tile_dim0
      g1[4] = 64;                                  // tile_dim1 = 64 rows, tile_dim2 = 0
      g1[5] = (int)td0;                            // tensor_dim0_stride lo32
      g1[6] = 0;                                   // stride0 hi16 | stride1 lo16
      g1[7] = 0;                                   // stride1 hi32
      v4i_t gz; gz[0] = 0; gz[1] = 0; gz[2] = 0; gz[3] = 0;
#if __clang_major__ >= 23
      v8i_t gz8; g1 = g1; gz8[0]=0; gz8[1]=0; gz8[2]=0; gz8[3]=0; gz8[4]=0; gz8[5]=0; gz8[6]=0; gz8[7]=0;
      __builtin_amdgcn_tensor_load_to_lds(g0, g1, gz, gz, gz8, 0);
#else
      __builtin_amdgcn_tensor_load_to_lds(g0, g1, gz, gz, 0);
#endif
      __builtin_amdgcn_s_wait_tensorcnt(0);
    }
#else
    {
      const __bf16* wsrc = Wt + btap * tap + (size_t)n_blk * CINP;
      const int nchunk = 64 * CINP / 8;            // 16B chunks
      for (int c = t; c < nchunk; c += 256) {
        int row = c / (CINP / 8), col = (c % (CINP / 8)) * 8;
        *(uint4*)(&Bs[row][col]) = *(const uint4*)(wsrc + (size_t)row * CINP + col);
      }
    }
#endif
    __syncthreads();                       // weight tile visible to all waves

    const int ih = aoh * strd + (tap / KW) - pad;
    const int iw = aow * strd + (tap % KW) - pad;
    const bool inb = am_ok && (unsigned)ih < (unsigned)IHd && (unsigned)iw < (unsigned)IWd;
    const __bf16* __restrict__ arowp =
        Xin + (((size_t)aimg * IHd + ih) * IWd + iw) * (size_t)CINP + akq;

#pragma unroll
    for (int k0 = 0; k0 < CINP; k0 += 32) {
      __syncthreads();
      uint4 av = {0u, 0u, 0u, 0u};
      if (inb) av = *(const uint4*)(arowp + k0);
      *(uint4*)(&As[arow][akq]) = av;
      __syncthreads();

      // Fragments per ISA 16-bit A/B layouts: both halves are 16B runs.
      v8bf_t alo = *(const v8bf_t*)(&As[mi * 16 + lr][half * 8]);
      v8bf_t ahi = *(const v8bf_t*)(&As[mi * 16 + lr][16 + half * 8]);
      v16bf_t af = __builtin_shufflevector(alo, ahi, 0,1,2,3,4,5,6,7,8,9,10,11,12,13,14,15);
      const int n0r = ni * 32 + lr;
      v8bf_t b0lo = *(const v8bf_t*)(&Bs[n0r][k0 + half * 8]);
      v8bf_t b0hi = *(const v8bf_t*)(&Bs[n0r][k0 + 16 + half * 8]);
      v8bf_t b1lo = *(const v8bf_t*)(&Bs[n0r + 16][k0 + half * 8]);
      v8bf_t b1hi = *(const v8bf_t*)(&Bs[n0r + 16][k0 + 16 + half * 8]);
      v16bf_t bf0 = __builtin_shufflevector(b0lo, b0hi, 0,1,2,3,4,5,6,7,8,9,10,11,12,13,14,15);
      v16bf_t bf1 = __builtin_shufflevector(b1lo, b1hi, 0,1,2,3,4,5,6,7,8,9,10,11,12,13,14,15);
      acc0 = __builtin_amdgcn_wmma_f32_16x16x32_bf16(false, af, false, bf0,
                                                     (short)0, acc0, false, false);
      acc1 = __builtin_amdgcn_wmma_f32_16x16x32_bf16(false, af, false, bf1,
                                                     (short)0, acc1, false, false);
    }
  }

  // Epilogue: C/D layout -> lane lr = N, VGPR r -> M = 8*half + r
  auto emit = [&](const v8f_t& a, int n) {
    const bool n_live = n < Nmax;
    float sc = 1.f, sh = 0.f, bi = 0.f;
    if (n_live) {
      if (flags & 1) bi = bias[n];
      if (flags & 2) { sc = bns[n]; sh = bnt[n]; }
    }
#pragma unroll
    for (int r = 0; r < 8; ++r) {
      const int m = m_blk + mi * 16 + half * 8 + r;
      if (m >= M) continue;
      float v = a[r] + bi;
      if (flags & 2) v = v * sc + sh;
      if (flags & 4) v = fmaxf(v, 0.f);
      if (SMODE == 0) {
        if (n < strideN) ((__bf16*)Out)[(size_t)m * strideN + n] = f2bf(n_live ? v : 0.f);
      } else if (SMODE == 1) {
        if (n_live) ((float*)Out)[(size_t)m * strideN + n] = v;
      } else {
        if (n_live) ((float*)Out)[(size_t)n * M + m] = v;
      }
    }
  };
  emit(acc0, n_blk + ni * 32 + lr);
  emit(acc1, n_blk + ni * 32 + lr + 16);
}

// ---------------------------------------------------------------------------
// Small kernels
// ---------------------------------------------------------------------------
static __device__ __forceinline__ void inv3(const float* m, float* o) {
  float a=m[0],b=m[1],c=m[2],d=m[3],e=m[4],f=m[5],g=m[6],h=m[7],i=m[8];
  float A=e*i-f*h, B=c*h-b*i, C=b*f-c*e;
  float D=f*g-d*i, E=a*i-c*g, F=c*d-a*f;
  float G=d*h-e*g, H=b*g-a*h, I=a*e-b*d;
  float det = a*A + b*D + c*G;
  float r = 1.f/det;
  o[0]=A*r; o[1]=B*r; o[2]=C*r;
  o[3]=D*r; o[4]=E*r; o[5]=F*r;
  o[6]=G*r; o[7]=H*r; o[8]=I*r;
}

__global__ void k_tf(const float* __restrict__ c2l_r, const float* __restrict__ c2l_t,
                     const float* __restrict__ intr, const float* __restrict__ post_r,
                     const float* __restrict__ post_t, float* __restrict__ tf) {
  int n = threadIdx.x;
  if (n >= BN_) return;
  float iPR[9], iK[9], C[9];
  inv3(post_r + n*9, iPR);
  inv3(intr + n*9, iK);
  for (int r = 0; r < 3; ++r)
    for (int c = 0; c < 3; ++c) {
      float s = 0.f;
      for (int k = 0; k < 3; ++k) s += c2l_r[n*9 + r*3 + k] * iK[k*3 + c];
      C[r*3+c] = s;
    }
  float* o = tf + n*24;
  for (int i = 0; i < 9; ++i) { o[i] = iPR[i]; o[9+i] = C[i]; }
  for (int i = 0; i < 3; ++i) { o[18+i] = c2l_t[n*3+i]; o[21+i] = post_t[n*3+i]; }
}

// Pack OIHW f32 weights -> [tap][O_pad][I_pad] bf16 (zero-padded)
__global__ void k_wpack(const float* __restrict__ w, __bf16* __restrict__ out,
                        int O, int I, int KHW, int O_pad, int I_pad) {
  long long total = (long long)KHW * O_pad * I_pad;
  long long i = (long long)blockIdx.x * blockDim.x + threadIdx.x;
  if (i >= total) return;
  int ci = (int)(i % I_pad);
  long long r = i / I_pad;
  int oc = (int)(r % O_pad);
  int tap = (int)(r / O_pad);
  float v = (oc < O && ci < I) ? w[((size_t)oc * I + ci) * KHW + tap] : 0.f;
  out[i] = f2bf(v);
}

// dt2 (5x5 s4 p2, 8->32) with dt1 (1x1 on raw depth) + both BN/ReLUs fused
__global__ void k_dt2(const float* __restrict__ d,
                      const float* __restrict__ w1, const float* __restrict__ b1,
                      const float* __restrict__ s1, const float* __restrict__ t1,
                      const float* __restrict__ w2, const float* __restrict__ b2,
                      const float* __restrict__ s2, const float* __restrict__ t2,
                      float* __restrict__ h2) {
  const int OH = 64, OW = 176;
  int i = blockIdx.x * blockDim.x + threadIdx.x;
  if (i >= BN_*32*OH*OW) return;
  int ow = i % OW; int t = i / OW;
  int oh = t % OH; t /= OH;
  int oc = t % 32; int img = t / 32;
  float A[8], Bc[8];
#pragma unroll
  for (int c = 0; c < 8; ++c) { A[c] = s1[c]*w1[c]; Bc[c] = s1[c]*b1[c] + t1[c]; }
  float acc = b2[oc];
  const float* dimg = d + (size_t)img * IH_ * IW_;
  const float* wocc = w2 + (size_t)oc * 8 * 25;
#pragma unroll
  for (int kh = 0; kh < 5; ++kh) {
    int ih = oh*4 + kh - 2;
    if ((unsigned)ih >= (unsigned)IH_) continue;
#pragma unroll
    for (int kw = 0; kw < 5; ++kw) {
      int iw = ow*4 + kw - 2;
      if ((unsigned)iw >= (unsigned)IW_) continue;
      float dv = dimg[(size_t)ih*IW_ + iw];
#pragma unroll
      for (int c = 0; c < 8; ++c) {
        float h1 = fmaxf(A[c]*dv + Bc[c], 0.f);
        acc += h1 * wocc[(c*5 + kh)*5 + kw];
      }
    }
  }
  h2[i] = fmaxf(acc * s2[oc] + t2[oc], 0.f);
}

// dt3 (5x5 s2 p2, 32->64) + BN/ReLU, writes bf16 into X1 cols [0,64)
__global__ void k_dt3(const float* __restrict__ h2,
                      const float* __restrict__ w3, const float* __restrict__ b3,
                      const float* __restrict__ s3, const float* __restrict__ t3,
                      __bf16* __restrict__ X1) {
  const int IH3 = 64, IW3 = 176;
  int i = blockIdx.x * blockDim.x + threadIdx.x;
  if (i >= BN_*64*FH_*FW_) return;
  int x = i % FW_; int t = i / FW_;
  int y = t % FH_; t /= FH_;
  int oc = t % 64; int img = t / 64;
  float acc = b3[oc];
  const float* wb = w3 + (size_t)oc*32*25;
  const float* hb = h2 + (size_t)img*32*IH3*IW3;
  for (int ci = 0; ci < 32; ++ci) {
    const float* hc = hb + (size_t)ci*IH3*IW3;
    const float* wc = wb + ci*25;
#pragma unroll
    for (int kh = 0; kh < 5; ++kh) {
      int ih = y*2 + kh - 2;
      if ((unsigned)ih >= (unsigned)IH3) continue;
#pragma unroll
      for (int kw = 0; kw < 5; ++kw) {
        int iw = x*2 + kw - 2;
        if ((unsigned)iw >= (unsigned)IW3) continue;
        acc += hc[ih*IW3 + iw] * wc[kh*5 + kw];
      }
    }
  }
  float v = fmaxf(acc*s3[oc] + t3[oc], 0.f);
  size_t m = (size_t)(img*FH_ + y)*FW_ + x;
  X1[m*320 + oc] = f2bf(v);
}

// x_img NCHW f32 -> X1 pixel-major bf16 cols [64,320)
__global__ void k_ximg(const float* __restrict__ xi, __bf16* __restrict__ X1) {
  int i = blockIdx.x * blockDim.x + threadIdx.x;
  if (i >= BN_*CIN_*FH_*FW_) return;
  int x = i % FW_; int t = i / FW_;
  int y = t % FH_; t /= FH_;
  int ci = t % CIN_; int img = t / CIN_;
  size_t m = (size_t)(img*FH_ + y)*FW_ + x;
  X1[m*320 + 64 + ci] = f2bf(xi[i]);
}

// In-place softmax over depth bins (cols 0..58 of F3, row stride 160)
__global__ void k_softmax(float* __restrict__ F3) {
  int m = blockIdx.x * blockDim.x + threadIdx.x;
  if (m >= NPIX_) return;
  float* row = F3 + (size_t)m * 160;
  float mx = row[0];
  for (int i = 1; i < DD_; ++i) mx = fmaxf(mx, row[i]);
  float s = 0.f;
  for (int i = 0; i < DD_; ++i) { float v = __expf(row[i] - mx); row[i] = v; s += v; }
  float r = 1.f / s;
  for (int i = 0; i < DD_; ++i) row[i] *= r;
}

// Frustum -> lidar -> voxel flat index (-1 if dropped)
__global__ void k_geom(const float* __restrict__ tf, int* __restrict__ idx) {
  int pt = blockIdx.x * blockDim.x + threadIdx.x;
  if (pt >= NPTS_) return;
  int w = pt % FW_; int t2 = pt / FW_;
  int h = t2 % FH_; t2 /= FH_;
  int dd = t2 % DD_; int n = t2 / DD_;
  const float* o = tf + n*24;
  float X = w * ((float)(IW_-1)/(FW_-1));
  float Y = h * ((float)(IH_-1)/(FH_-1));
  float Z = 1.f + (float)dd;
  float p0 = X - o[21], p1 = Y - o[22], p2 = Z - o[23];
  float q0 = o[0]*p0 + o[1]*p1 + o[2]*p2;
  float q1 = o[3]*p0 + o[4]*p1 + o[5]*p2;
  float q2 = o[6]*p0 + o[7]*p1 + o[8]*p2;
  q0 *= q2; q1 *= q2;
  float r0 = o[9]*q0  + o[10]*q1 + o[11]*q2 + o[18];
  float r1 = o[12]*q0 + o[13]*q1 + o[14]*q2 + o[19];
  float r2 = o[15]*q0 + o[16]*q1 + o[17]*q2 + o[20];
  int gx = (int)((r0 + 54.f) / 0.3f);
  int gy = (int)((r1 + 54.f) / 0.3f);
  int gz = (int)((r2 + 10.f) / 20.f);
  bool kept = gx >= 0 && gx < NXG_ && gy >= 0 && gy < NXG_ && gz == 0;
  idx[pt] = kept ? (gx * NXG_ + gy) : -1;
}

__global__ void k_zero(float* __restrict__ p, int n) {
  int i = blockIdx.x * blockDim.x + threadIdx.x;
  if (i < n) p[i] = 0.f;
}

// bev_pool scatter: per (point, 16-channel group) -> fp32 atomics (L2 resident)
__global__ void k_scatter(const float* __restrict__ F3, const int* __restrict__ idx,
                          float* __restrict__ bev) {
  int tid = blockIdx.x * blockDim.x + threadIdx.x;
  if (tid >= NPTS_ * 5) return;
  int pt = tid / 5, cg = tid % 5;
  int fl = idx[pt];
  if (fl < 0) return;
  int w = pt % FW_; int t2 = pt / FW_;
  int h = t2 % FH_; t2 /= FH_;
  int dd = t2 % DD_; int n = t2 / DD_;
  size_t m = (size_t)(n * FH_ + h) * FW_ + w;
  const float* row = F3 + m * 160;
  float p = row[dd];
  const float* fp = row + DD_ + cg * 16;
  float* dst = bev + (size_t)fl * CIMG_ + cg * 16;
#pragma unroll
  for (int c = 0; c < 16; ++c) atomicAdd(dst + c, p * fp[c]);
}

// BEV f32 [p][80] -> bf16 [p][96] (zero pad cols)
__global__ void k_xbpack(const float* __restrict__ bev, __bf16* __restrict__ xb) {
  int i = blockIdx.x * blockDim.x + threadIdx.x;
  if (i >= NXG_*NXG_*96) return;
  int c = i % 96; int p = i / 96;
  xb[i] = f2bf(c < CIMG_ ? bev[(size_t)p * CIMG_ + c] : 0.f);
}

// ---------------------------------------------------------------------------
// Host launcher
// ---------------------------------------------------------------------------
extern "C" void kernel_launch(void* const* d_in, const int* in_sizes, int n_in,
                              void* d_out, int out_size, void* d_ws, size_t ws_size,
                              hipStream_t stream) {
  (void)in_sizes; (void)n_in; (void)out_size; (void)ws_size;
  const float* x_img = (const float*)d_in[0];
  const float* dmap  = (const float*)d_in[1];
  const float* c2l_r = (const float*)d_in[2];
  const float* c2l_t = (const float*)d_in[3];
  const float* intr  = (const float*)d_in[4];
  const float* postr = (const float*)d_in[5];
  const float* postt = (const float*)d_in[6];
  const float* dt1_w = (const float*)d_in[7];
  const float* dt1_b = (const float*)d_in[8];
  const float* dt1_s = (const float*)d_in[9];
  const float* dt1_t = (const float*)d_in[10];
  const float* dt2_w = (const float*)d_in[11];
  const float* dt2_b = (const float*)d_in[12];
  const float* dt2_s = (const float*)d_in[13];
  const float* dt2_t = (const float*)d_in[14];
  const float* dt3_w = (const float*)d_in[15];
  const float* dt3_b = (const float*)d_in[16];
  const float* dt3_s = (const float*)d_in[17];
  const float* dt3_t = (const float*)d_in[18];
  const float* dn1_w = (const float*)d_in[19];
  const float* dn1_b = (const float*)d_in[20];
  const float* dn1_s = (const float*)d_in[21];
  const float* dn1_t = (const float*)d_in[22];
  const float* dn2_w = (const float*)d_in[23];
  const float* dn2_b = (const float*)d_in[24];
  const float* dn2_s = (const float*)d_in[25];
  const float* dn2_t = (const float*)d_in[26];
  const float* dn3_w = (const float*)d_in[27];
  const float* dn3_b = (const float*)d_in[28];
  const float* ds1_w = (const float*)d_in[29];
  const float* ds1_s = (const float*)d_in[30];
  const float* ds1_t = (const float*)d_in[31];
  const float* ds2_w = (const float*)d_in[32];
  const float* ds2_s = (const float*)d_in[33];
  const float* ds2_t = (const float*)d_in[34];

  char* ws = (char*)d_ws;
  size_t off = 0;
  auto alloc = [&](size_t bytes) { size_t o = off; off += (bytes + 255) & ~(size_t)255; return o; };

  float*  TF   = (float*) (ws + alloc((size_t)BN_*24*4));
  float*  H2   = (float*) (ws + alloc((size_t)BN_*32*64*176*4));
  __bf16* X1   = (__bf16*)(ws + alloc((size_t)NPIX_*320*2));
  __bf16* X2   = (__bf16*)(ws + alloc((size_t)NPIX_*256*2));
  __bf16* W1t  = (__bf16*)(ws + alloc((size_t)9*256*320*2));
  __bf16* W2t  = (__bf16*)(ws + alloc((size_t)9*256*256*2));
  __bf16* W3t  = (__bf16*)(ws + alloc((size_t)1*192*256*2));
  float*  F3   = (float*) (ws + alloc((size_t)NPIX_*160*4));
  int*    IDX  = (int*)   (ws + alloc((size_t)NPTS_*4));
  float*  BEV  = (float*) (ws + alloc((size_t)NXG_*NXG_*CIMG_*4));
  __bf16* XB   = (__bf16*)(ws + alloc((size_t)NXG_*NXG_*96*2));
  __bf16* WD1t = (__bf16*)(ws + alloc((size_t)9*128*96*2));
  __bf16* WD2t = (__bf16*)(ws + alloc((size_t)9*128*96*2));
  __bf16* Y1   = (__bf16*)(ws + alloc((size_t)180*180*96*2));

  auto blks = [](long long n) { return (unsigned)((n + 255) / 256); };

  // transforms + weight packing (input-dependent -> every call)
  k_tf<<<1, 32, 0, stream>>>(c2l_r, c2l_t, intr, postr, postt, TF);
  k_wpack<<<blks((long long)9*256*320), 256, 0, stream>>>(dn1_w, W1t, 256, 320, 9, 256, 320);
  k_wpack<<<blks((long long)9*256*256), 256, 0, stream>>>(dn2_w, W2t, 256, 256, 9, 256, 256);
  k_wpack<<<blks((long long)1*192*256), 256, 0, stream>>>(dn3_w, W3t, 139, 256, 1, 192, 256);
  k_wpack<<<blks((long long)9*128*96),  256, 0, stream>>>(ds1_w, WD1t, 80, 80, 9, 128, 96);
  k_wpack<<<blks((long long)9*128*96),  256, 0, stream>>>(ds2_w, WD2t, 80, 80, 9, 128, 96);

  // dtransform (dt1 fused into dt2) + concat assembly into X1
  k_dt2<<<blks((long long)BN_*32*64*176), 256, 0, stream>>>(
      dmap, dt1_w, dt1_b, dt1_s, dt1_t, dt2_w, dt2_b, dt2_s, dt2_t, H2);
  k_dt3<<<blks((long long)BN_*64*FH_*FW_), 256, 0, stream>>>(
      H2, dt3_w, dt3_b, dt3_s, dt3_t, X1);
  k_ximg<<<blks((long long)BN_*CIN_*FH_*FW_), 256, 0, stream>>>(x_img, X1);

  // depthnet: dn1, dn2 (bf16 chained), dn3 (f32 -> F3)
  {
    dim3 g(NPIX_ / 64, 4);                    // 264 x 4
    conv_wmma<0, 320><<<g, 256, 0, stream>>>(X1, W1t, X2, dn1_b, dn1_s, dn1_t, 7,
                                             FH_, FW_, FH_, FW_, 3, 3, 1, 1,
                                             256, 256, 256, NPIX_);
    conv_wmma<0, 256><<<g, 256, 0, stream>>>(X2, W2t, X1, dn2_b, dn2_s, dn2_t, 7,
                                             FH_, FW_, FH_, FW_, 3, 3, 1, 1,
                                             256, 256, 256, NPIX_);
    dim3 g3(NPIX_ / 64, 3);                   // 264 x 3
    conv_wmma<1, 256><<<g3, 256, 0, stream>>>(X1, W3t, F3, dn3_b, nullptr, nullptr, 1,
                                              FH_, FW_, FH_, FW_, 1, 1, 1, 0,
                                              192, 139, 160, NPIX_);
  }

  // softmax over depth bins (in place in F3)
  k_softmax<<<blks(NPIX_), 256, 0, stream>>>(F3);

  // geometry + bev_pool
  k_geom<<<blks(NPTS_), 256, 0, stream>>>(TF, IDX);
  k_zero<<<blks((long long)NXG_*NXG_*CIMG_), 256, 0, stream>>>(BEV, NXG_*NXG_*CIMG_);
  k_scatter<<<blks((long long)NPTS_*5), 256, 0, stream>>>(F3, IDX, BEV);

  // downsample: pack BEV -> bf16, ds1 (bf16 out), ds2 (f32 NCHW -> d_out)
  k_xbpack<<<blks((long long)NXG_*NXG_*96), 256, 0, stream>>>(BEV, XB);
  {
    dim3 g((32400 + 63) / 64, 2);             // 507 x 2
    conv_wmma<0, 96><<<g, 256, 0, stream>>>(XB, WD1t, Y1, nullptr, ds1_s, ds1_t, 6,
                                            NXG_, NXG_, 180, 180, 3, 3, 2, 1,
                                            128, 80, 96, 32400);
    conv_wmma<2, 96><<<g, 256, 0, stream>>>(Y1, WD2t, d_out, nullptr, ds2_s, ds2_t, 6,
                                            180, 180, 180, 180, 3, 3, 1, 1,
                                            128, 80, 0, 32400);
  }
}
